// Attention_8366596292664
// MI455X (gfx1250) — matrix operand order repaired and checked by reading, not compiled, not run on hardware
//
#include <hip/hip_runtime.h>
#include <hip/hip_bf16.h>
#include <cstdint>
#include <cstddef>

// ---------------- problem constants ----------------
constexpr int Bc   = 4;
constexpr int S    = 2048;
constexpr int H    = 2048;
constexpr int NH   = 16;
constexpr int HD   = 128;          // head dim
constexpr int G    = 4;            // GQA group
constexpr int NKV  = NH / G;       // 4 kv heads
constexpr int GHD  = H / G;        // 512
constexpr int MROWS = Bc * S;      // 8192
constexpr int NQKV = H + 2 * GHD;  // 3072

// workspace layout (bytes)
constexpr size_t OFF_QKV = 0;                                       // f32 [8192][3072]
constexpr size_t SZ_QKV  = (size_t)MROWS * NQKV * 4;
constexpr size_t OFF_QB  = OFF_QKV + SZ_QKV;                        // bf16 [B][NH][S][HD]
constexpr size_t SZ_QB   = (size_t)Bc * NH * S * HD * 2;
constexpr size_t OFF_KB  = OFF_QB + SZ_QB;                          // bf16 [B][NKV][S][HD]
constexpr size_t SZ_KB   = (size_t)Bc * NKV * S * HD * 2;
constexpr size_t OFF_VB  = OFF_KB + SZ_KB;
constexpr size_t SZ_VB   = SZ_KB;
constexpr size_t OFF_YB  = OFF_VB + SZ_VB;                          // bf16 [8192][2048]
constexpr size_t SZ_YB   = (size_t)MROWS * H * 2;
constexpr size_t OFF_YP  = OFF_YB + SZ_YB;                          // f32 [8192][2048]

#define GLOBAL_AS __attribute__((address_space(1)))
#define LOCAL_AS  __attribute__((address_space(3)))

// -------- CDNA5 feature detection (compile-safe) --------
#if __has_builtin(__builtin_amdgcn_global_load_async_to_lds_b128) && \
    __has_builtin(__builtin_amdgcn_s_wait_asynccnt)
#define HAVE_ASYNC 1
#else
#define HAVE_ASYNC 0
#endif

#if __has_builtin(__builtin_amdgcn_tensor_load_to_lds) && \
    __has_builtin(__builtin_amdgcn_s_wait_tensorcnt)
#define HAVE_TDM 1
#else
#define HAVE_TDM 0
#endif

// ---------------- WMMA / vector types ----------------
typedef __attribute__((ext_vector_type(16))) __bf16 v16bf;
typedef __attribute__((ext_vector_type(8)))  float  v8f;
typedef __attribute__((ext_vector_type(4))) unsigned v4u;
typedef __attribute__((ext_vector_type(8))) int      v8i;
typedef __attribute__((ext_vector_type(4))) int      v4i;

__device__ inline unsigned short f32_to_bf16(float f) {
    unsigned u = __builtin_bit_cast(unsigned, f);
    unsigned r = u + 0x7FFFu + ((u >> 16) & 1u);   // round-to-nearest-even
    return (unsigned short)(r >> 16);
}
__device__ inline unsigned pack_bf16x2(float a, float b) {
    return (unsigned)f32_to_bf16(a) | ((unsigned)f32_to_bf16(b) << 16);
}
__device__ inline v16bf make_frag(const unsigned u[8]) {
    union { unsigned x[8]; v16bf v; } t;
#pragma unroll
    for (int i = 0; i < 8; ++i) t.x[i] = u[i];
    return t.v;
}
__device__ inline v8f wmma_bf16(v16bf a, v16bf b, v8f c) {
    return __builtin_amdgcn_wmma_f32_16x16x32_bf16(false, a, false, b,
                                                   (short)0, c, false, false);
}
__device__ inline v8f v8f_zero() {
    v8f z;
#pragma unroll
    for (int i = 0; i < 8; ++i) z[i] = 0.0f;
    return z;
}

#if HAVE_ASYNC
__device__ inline void async_copy_b128(const void* gsrc, void* ldst) {
    void* gp = (void*)gsrc;   // drop const; builtin takes non-const vector pointers
    __builtin_amdgcn_global_load_async_to_lds_b128(
        (GLOBAL_AS v4i*)gp, (LOCAL_AS v4i*)ldst, 0, 0);
}
#endif

#if HAVE_TDM
// TDM descriptor for a 128x32 bf16 tile of A[MROWS][K] -> LDS As[128][17] uints
// (pad_interval=16 DWORDs, pad_amount=1 DWORD reproduces the 17-uint padded row).
__device__ inline void tdm_load_a_tile(const void* Ap, int mBase, int k0, int K,
                                       unsigned lds_off) {
    const unsigned long long ga =
        (unsigned long long)(uintptr_t)Ap + ((size_t)mBase * K + k0) * 2ull;
    v4u g0;
    g0[0] = 1u;                                            // count=1 user descriptor
    g0[1] = lds_off;                                       // lds_addr
    g0[2] = (unsigned)(ga & 0xFFFFFFFFu);                  // global_addr[31:0]
    g0[3] = (unsigned)((ga >> 32) & 0x01FFFFFFu) | (2u << 30);  // addr[56:32], type=2
    v8i g1;
    g1[0] = (int)((1u << 16) | (1u << 20) | (3u << 22));   // data_size=2B, pad_en, 16DW interval, 1DW pad
    g1[1] = (int)((unsigned)(K & 0xFFFF) << 16);           // tensor_dim0[15:0]
    g1[2] = (int)(((unsigned)(MROWS & 0xFFFF) << 16) | ((unsigned)(K >> 16) & 0xFFFFu)); // dim1 lo | dim0 hi
    g1[3] = (int)(32u << 16);                              // tile_dim0=32 (dim1 hi = 0)
    g1[4] = 128;                                           // tile_dim1=128, tile_dim2=0
    g1[5] = K;                                             // tensor_dim0_stride lo32
    g1[6] = 0;
    g1[7] = 0;
    v4i z4; z4[0] = z4[1] = z4[2] = z4[3] = 0;
#if __clang_major__ >= 23
    v8i z8;
#pragma unroll
    for (int i = 0; i < 8; ++i) z8[i] = 0;
    __builtin_amdgcn_tensor_load_to_lds(g0, g1, z4, z4, z8, 0);
#else
    __builtin_amdgcn_tensor_load_to_lds(g0, g1, z4, z4, 0);
#endif
}
#endif

// ---------------- GEMM: C[M,N] f32 = A[M,K] (f32 or bf16) * W[K,N] f32 ----------------
// 128x128 block tile, BK=32, 256 threads = 8 waves (4x2), register-staged pipeline.
// When A is bf16 and TDM is available, the A tile is DMA'd by the Tensor Data Mover.
template <bool ABF16>
__global__ __launch_bounds__(256) void gemm_bf16_kernel(const void* __restrict__ Ap,
                                                        const float* __restrict__ W,
                                                        float* __restrict__ C,
                                                        int N, int K) {
    __shared__ unsigned As[128][17];   // [m][k-pair] (16 uints + 1 pad) — TDM-compatible
    __shared__ unsigned Bs[128][17];   // [n][k-pair]

    const int tid   = threadIdx.x;
    const int mBase = blockIdx.y * 128;
    const int nBase = blockIdx.x * 128;
    const int wave  = tid >> 5, lane = tid & 31;
    const int hl    = lane >> 4, ln = lane & 15;
    const int wm    = wave & 3;
    const int wn    = wave >> 2;

#if HAVE_TDM
    constexpr bool kTdmA = ABF16;
    const unsigned asLdsOff = (unsigned)(uintptr_t)(LOCAL_AS void*)&As[0][0];
#else
    constexpr bool kTdmA = false;
#endif

    v8f acc[2][4];
#pragma unroll
    for (int mt = 0; mt < 2; ++mt)
#pragma unroll
        for (int nt = 0; nt < 4; ++nt) acc[mt][nt] = v8f_zero();

    const int tRow = tid >> 3;         // 0..31
    const int tK   = (tid & 7) * 4;    // elements
    const int tn   = tid & 127;
    const int tkq  = tid >> 7;

    // register staging for the next K-slab
    float4 aF[4];
    uint2  aB[4];
    float  w0r[8], w1r[8];

    auto load_regs = [&](int k0) {
        if (!kTdmA) {
#pragma unroll
            for (int i = 0; i < 4; ++i) {
                const int m = i * 32 + tRow;
                if (ABF16) {
                    const unsigned short* a16 = (const unsigned short*)Ap;
                    aB[i] = *(const uint2*)(a16 + (size_t)(mBase + m) * K + k0 + tK);
                } else {
                    aF[i] = *(const float4*)((const float*)Ap + (size_t)(mBase + m) * K + k0 + tK);
                }
            }
        }
#pragma unroll
        for (int j = 0; j < 8; ++j) {
            const int k = k0 + 2 * (tkq * 8 + j);
            w0r[j] = W[(size_t)k * N + nBase + tn];
            w1r[j] = W[(size_t)(k + 1) * N + nBase + tn];
        }
        if (k0 + 32 < K)
            __builtin_prefetch(&W[(size_t)(k0 + 32) * N + nBase + tn], 0, 1);
    };

    load_regs(0);
    for (int k0 = 0; k0 < K; k0 += 32) {
#if HAVE_TDM
        if (kTdmA && wave == 0) tdm_load_a_tile(Ap, mBase, k0, K, asLdsOff);
#endif
        // commit staged registers to LDS
        if (!kTdmA) {
#pragma unroll
            for (int i = 0; i < 4; ++i) {
                const int m = i * 32 + tRow;
                if (ABF16) {
                    As[m][(tK >> 1) + 0] = aB[i].x;
                    As[m][(tK >> 1) + 1] = aB[i].y;
                } else {
                    As[m][(tK >> 1) + 0] = pack_bf16x2(aF[i].x, aF[i].y);
                    As[m][(tK >> 1) + 1] = pack_bf16x2(aF[i].z, aF[i].w);
                }
            }
        }
#pragma unroll
        for (int j = 0; j < 8; ++j)
            Bs[tn][tkq * 8 + j] = pack_bf16x2(w0r[j], w1r[j]);
#if HAVE_TDM
        if (kTdmA && wave == 0) __builtin_amdgcn_s_wait_tensorcnt(0);
#endif
        __syncthreads();

        if (k0 + 32 < K) load_regs(k0 + 32);   // overlap next-slab loads with WMMA

        unsigned af[2][8], bfr[4][8];
#pragma unroll
        for (int mt = 0; mt < 2; ++mt) {
            const int row = wm * 32 + mt * 16 + ln;
#pragma unroll
            for (int v = 0; v < 8; ++v)
                af[mt][v] = As[row][(v >> 2) * 8 + hl * 4 + (v & 3)];
        }
#pragma unroll
        for (int nt = 0; nt < 4; ++nt) {
            const int col = wn * 64 + nt * 16 + ln;
#pragma unroll
            for (int v = 0; v < 8; ++v)
                bfr[nt][v] = Bs[col][(v >> 2) * 8 + hl * 4 + (v & 3)];
        }
#pragma unroll
        for (int mt = 0; mt < 2; ++mt) {
            const v16bf a = make_frag(af[mt]);
#pragma unroll
            for (int nt = 0; nt < 4; ++nt)
                acc[mt][nt] = wmma_bf16(a, make_frag(bfr[nt]), acc[mt][nt]);
        }
        __syncthreads();
    }

#pragma unroll
    for (int mt = 0; mt < 2; ++mt)
#pragma unroll
        for (int nt = 0; nt < 4; ++nt)
#pragma unroll
            for (int r = 0; r < 8; ++r) {
                const int row = mBase + wm * 32 + mt * 16 + r + 8 * hl;
                const int col = nBase + wn * 64 + nt * 16 + ln;
                C[(size_t)row * N + col] = acc[mt][nt][r];
            }
}

// ---------------- RoPE + split + transpose + bf16 pack ----------------
__global__ __launch_bounds__(256) void rope_pack_kernel(const float* __restrict__ qkv,
                                                        unsigned short* __restrict__ qb,
                                                        unsigned short* __restrict__ kb,
                                                        unsigned short* __restrict__ vb) {
    const int m = blockIdx.x;
    const int b = m >> 11, s = m & 2047;
    const float* row = qkv + (size_t)m * NQKV;
    for (int p = threadIdx.x; p < NQKV / 2; p += 256) {
        const int f = p * 2;
        if (f < H) {                         // Q with RoPE
            const int hh = f >> 7, d = f & 127;
            const float ang = (float)s * __powf(10000.0f, -(float)(d) / 128.0f);
            float sn, cs; __sincosf(ang, &sn, &cs);
            const float x0 = row[f], x1 = row[f + 1];
            unsigned* dst = (unsigned*)(qb + ((size_t)(b * NH + hh) * S + s) * HD + d);
            *dst = pack_bf16x2(x0 * cs - x1 * sn, x0 * sn + x1 * cs);
        } else if (f < H + GHD) {            // K with RoPE
            const int ff = f - H;
            const int kh = ff >> 7, d = ff & 127;
            const float ang = (float)s * __powf(10000.0f, -(float)(d) / 128.0f);
            float sn, cs; __sincosf(ang, &sn, &cs);
            const float x0 = row[f], x1 = row[f + 1];
            unsigned* dst = (unsigned*)(kb + ((size_t)(b * NKV + kh) * S + s) * HD + d);
            *dst = pack_bf16x2(x0 * cs - x1 * sn, x0 * sn + x1 * cs);
        } else {                             // V passthrough
            const int ff = f - H - GHD;
            const int vh = ff >> 7, d = ff & 127;
            unsigned* dst = (unsigned*)(vb + ((size_t)(b * NKV + vh) * S + s) * HD + d);
            *dst = pack_bf16x2(row[f], row[f + 1]);
        }
    }
}

// ---------------- causal GQA flash attention (bf16 WMMA, f32 softmax) ----------------
// grid (S/64, NH, B), 128 threads = 4 waves, each wave owns 16 q rows.
// K/V tiles are double-buffered in LDS and fed by global_load_async_to_lds_b128,
// pipelined one kv-block ahead with s_wait_asynccnt.
__global__ __launch_bounds__(128) void flash_attn_kernel(const unsigned short* __restrict__ qb,
                                                         const unsigned short* __restrict__ kb,
                                                         const unsigned short* __restrict__ vb,
                                                         unsigned short* __restrict__ yb) {
    // 272B row stride (256B data + 16B pad) keeps fragment gathers spread over banks
    __shared__ __attribute__((aligned(16))) unsigned char KsRaw[2][32][272];
    __shared__ __attribute__((aligned(16))) unsigned char VrRaw[2][32][272];
    __shared__ unsigned short Ps[4][16][34];   // per-wave P staging

    const int b = blockIdx.z, h = blockIdx.y, q0 = blockIdx.x * 64;
    const int kvh = h >> 2;
    const int tid = threadIdx.x, wave = tid >> 5, lane = tid & 31;
    const int hl = lane >> 4, ln = lane & 15;

    const unsigned short* Qbase = qb + (size_t)(b * NH + h) * S * HD;
    const unsigned short* Kbase = kb + (size_t)(b * NKV + kvh) * S * HD;
    const unsigned short* Vbase = vb + (size_t)(b * NKV + kvh) * S * HD;

    // Q fragments: 4 chunks of K=32 over head dim, kept in registers
    unsigned qf[4][8];
    {
        const int qrow = q0 + wave * 16 + ln;
        const unsigned short* qptr = Qbase + (size_t)qrow * HD;
#pragma unroll
        for (int c = 0; c < 4; ++c)
#pragma unroll
            for (int v = 0; v < 8; ++v) {
                const int Kd = c * 32 + (v >> 2) * 16 + hl * 8 + (v & 3) * 2;
                qf[c][v] = *(const unsigned*)(qptr + Kd);
            }
    }

    v8f o[8];
#pragma unroll
    for (int dt = 0; dt < 8; ++dt) o[dt] = v8f_zero();
    float mstate[8], lstate[8];
#pragma unroll
    for (int r = 0; r < 8; ++r) { mstate[r] = -1e30f; lstate[r] = 0.0f; }

    const float scale = 0.08838834764831845f;  // 1/sqrt(128)
    const int jmax = (q0 + 64) >> 5;

#if HAVE_ASYNC
    // each wave DMAs 8 x 512B (4 K-chunks + 4 V-chunks) per kv-block
    auto issue_async = [&](int j, int buf) {
#pragma unroll
        for (int i = 0; i < 4; ++i) {
            const int chunk = wave * 128 + i * 32 + lane;   // 0..511
            const int key = chunk >> 4, cb = chunk & 15;
            const size_t gOff = (size_t)(j * 32 + key) * 256 + cb * 16;
            async_copy_b128((const char*)Kbase + gOff, &KsRaw[buf][key][cb * 16]);
            async_copy_b128((const char*)Vbase + gOff, &VrRaw[buf][key][cb * 16]);
        }
    };
#else
    auto stage_sync = [&](int j) {
#pragma unroll
        for (int i = 0; i < 4; ++i) {
            const int chunk = tid * 4 + i;
            const int key = chunk >> 4, cb = chunk & 15;
            const size_t gOff = (size_t)(j * 32 + key) * 256 + cb * 16;
            *(uint4*)&KsRaw[0][key][cb * 16] = *(const uint4*)((const char*)Kbase + gOff);
            *(uint4*)&VrRaw[0][key][cb * 16] = *(const uint4*)((const char*)Vbase + gOff);
        }
    };
#endif

    auto compute_block = [&](int j, int cur) {
        const int jb = j * 32;
        // ---- scores: S(16q x 32k) over 4 d-chunks ----
        v8f sacc[2]; sacc[0] = v8f_zero(); sacc[1] = v8f_zero();
#pragma unroll
        for (int nt = 0; nt < 2; ++nt) {
            const unsigned* krow = (const unsigned*)&KsRaw[cur][nt * 16 + ln][0];
#pragma unroll
            for (int c = 0; c < 4; ++c) {
                unsigned kf[8];
#pragma unroll
                for (int v = 0; v < 8; ++v)
                    kf[v] = krow[c * 16 + (v >> 2) * 8 + hl * 4 + (v & 3)];
                sacc[nt] = wmma_bf16(make_frag(qf[c]), make_frag(kf), sacc[nt]);
            }
        }
        // ---- online softmax per row ----
#pragma unroll
        for (int r = 0; r < 8; ++r) {
            const int qi = q0 + wave * 16 + r + 8 * hl;
            float s0 = sacc[0][r] * scale;
            float s1 = sacc[1][r] * scale;
            if (jb + ln > qi)      s0 = -1e30f;
            if (jb + 16 + ln > qi) s1 = -1e30f;
            float rm = fmaxf(s0, s1);
#pragma unroll
            for (int msk = 8; msk >= 1; msk >>= 1) rm = fmaxf(rm, __shfl_xor(rm, msk, 32));
            const float mnew = fmaxf(mstate[r], rm);
            const float corr = __expf(mstate[r] - mnew);
            const float p0 = __expf(s0 - mnew);
            const float p1 = __expf(s1 - mnew);
            float rs = p0 + p1;
#pragma unroll
            for (int msk = 8; msk >= 1; msk >>= 1) rs += __shfl_xor(rs, msk, 32);
            lstate[r] = lstate[r] * corr + rs;
            mstate[r] = mnew;
#pragma unroll
            for (int dt = 0; dt < 8; ++dt) o[dt][r] *= corr;
            Ps[wave][r + 8 * hl][ln]      = f32_to_bf16(p0);
            Ps[wave][r + 8 * hl][16 + ln] = f32_to_bf16(p1);
        }
        // ---- O += P(16x32) * V(32x128) ----
        unsigned pf[8];
#pragma unroll
        for (int v = 0; v < 8; ++v)
            pf[v] = ((const unsigned*)&Ps[wave][ln][0])[hl * 8 + v];
        const v16bf pa = make_frag(pf);
#pragma unroll
        for (int dt = 0; dt < 8; ++dt) {
            const int d = dt * 16 + ln;
            unsigned vf[8];
#pragma unroll
            for (int v = 0; v < 8; ++v) {
                const unsigned lo = *(const unsigned short*)&VrRaw[cur][hl * 16 + 2 * v][d * 2];
                const unsigned hi = *(const unsigned short*)&VrRaw[cur][hl * 16 + 2 * v + 1][d * 2];
                vf[v] = lo | (hi << 16);
            }
            o[dt] = wmma_bf16(pa, make_frag(vf), o[dt]);
        }
    };

#if HAVE_ASYNC
    issue_async(0, 0);
    for (int j = 0; j < jmax; ++j) {
        const int cur = j & 1;
        if (j + 1 < jmax) {
            issue_async(j + 1, cur ^ 1);           // prefetch next kv-block
            __builtin_amdgcn_s_wait_asynccnt(8);   // this block's 8 DMAs are done (in-order)
        } else {
            __builtin_amdgcn_s_wait_asynccnt(0);
        }
        __syncthreads();
        compute_block(j, cur);
        __syncthreads();
    }
#else
    for (int j = 0; j < jmax; ++j) {
        stage_sync(j);
        __syncthreads();
        compute_block(j, 0);
        __syncthreads();
    }
#endif

    // epilogue: O /= l, write bf16 into (B,S,H) layout for the projection GEMM
#pragma unroll
    for (int dt = 0; dt < 8; ++dt)
#pragma unroll
        for (int r = 0; r < 8; ++r) {
            const int qi = q0 + wave * 16 + r + 8 * hl;
            const float val = o[dt][r] / lstate[r];
            yb[(size_t)(b * S + qi) * H + h * HD + dt * 16 + ln] = f32_to_bf16(val);
        }
}

// ---------------- RMSNorm ----------------
__global__ __launch_bounds__(256) void rmsnorm_kernel(const float* __restrict__ y,
                                                      const float* __restrict__ w,
                                                      float* __restrict__ out) {
    __shared__ float red[256];
    const int m = blockIdx.x;
    const int tid = threadIdx.x;
    const float* row = y + (size_t)m * H;
    float ss = 0.0f;
    for (int c = tid; c < H; c += 256) { const float v = row[c]; ss += v * v; }
    red[tid] = ss;
    __syncthreads();
    for (int s = 128; s > 0; s >>= 1) {
        if (tid < s) red[tid] += red[tid + s];
        __syncthreads();
    }
    const float rs = rsqrtf(red[0] / (float)H + 1e-6f);
    for (int c = tid; c < H; c += 256)
        out[(size_t)m * H + c] = row[c] * rs * w[c];
}

// ---------------- host launcher ----------------
extern "C" void kernel_launch(void* const* d_in, const int* in_sizes, int n_in,
                              void* d_out, int out_size, void* d_ws, size_t ws_size,
                              hipStream_t stream) {
    const float* x      = (const float*)d_in[0];
    const float* w_attn = (const float*)d_in[1];
    const float* w_proj = (const float*)d_in[2];
    const float* norm_w = (const float*)d_in[3];
    float* out = (float*)d_out;

    char* ws = (char*)d_ws;
    float*          qkv   = (float*)(ws + OFF_QKV);
    unsigned short* qb    = (unsigned short*)(ws + OFF_QB);
    unsigned short* kb    = (unsigned short*)(ws + OFF_KB);
    unsigned short* vb    = (unsigned short*)(ws + OFF_VB);
    unsigned short* yb    = (unsigned short*)(ws + OFF_YB);
    float*          yproj = (float*)(ws + OFF_YP);

    gemm_bf16_kernel<false><<<dim3(NQKV / 128, MROWS / 128), 256, 0, stream>>>(
        (const void*)x, w_attn, qkv, NQKV, H);
    rope_pack_kernel<<<MROWS, 256, 0, stream>>>(qkv, qb, kb, vb);
    flash_attn_kernel<<<dim3(S / 64, NH, Bc), 128, 0, stream>>>(qb, kb, vb, yb);
    gemm_bf16_kernel<true><<<dim3(H / 128, MROWS / 128), 256, 0, stream>>>(
        (const void*)yb, w_proj, yproj, H, H);
    rmsnorm_kernel<<<MROWS, 256, 0, stream>>>(yproj, norm_w, out);
}
